// PriorLayer_16329465659709
// MI455X (gfx1250) — matrix operand order repaired and checked by reading, not compiled, not run on hardware
//
#include <hip/hip_runtime.h>
#include <math.h>

// HMM forward (sum-product belief propagation), DIM=256, gfx1250.
//
// Serial chain of 131072 dependent matvec+normalize steps -> latency bound,
// one workgroup (8 wave32) owns the chain. T is resident in per-wave WMMA
// B-operand registers (f16, 128 VGPRs/wave). The recursion carries an
// UNNORMALIZED state v_i scaled by 65536/s_{i-1} (s = measured sum of the
// previous state), so the sum-reduction, rcp, probs store and std all lag one
// step and run OFF the critical path, overlapped with the WMMA phase:
//   XDL pipe : 16 x v_wmma_f32_16x16x32_f16  (T @ v, 2 output blocks/wave)
//   DS  pipe : 15 x ds_bpermute butterfly (3 sums of v_{i-1}) + red exchange
// State crosses waves via double-buffered f16 LDS; 2 barriers/step.

typedef __attribute__((ext_vector_type(16))) _Float16 v16h;
typedef __attribute__((ext_vector_type(8)))  _Float16 v8h;
typedef __attribute__((ext_vector_type(8)))  float    v8f;
typedef __attribute__((ext_vector_type(4)))  float    v4f;

#define DIM    256
#define WAVES  8
#define PF     8          // ps register-prefetch depth (iterations)
#define PF_L2  48         // global_prefetch lookahead (iterations)
#define VSCALE 65536.0f   // target sum of the carried unnormalized state

static __device__ __forceinline__ float fast_rcp(float x) {
#if __has_builtin(__builtin_amdgcn_rcpf)
    return __builtin_amdgcn_rcpf(x);   // 1-ulp; scale error cancels exactly
#else
    return 1.0f / x;
#endif
}

__global__ __launch_bounds__(256, 1)
void hmm_forward_wmma(const float* __restrict__ ps,      // [n, 256]
                      const float* __restrict__ T,       // [256, 256]
                      const float* __restrict__ state0,  // [256]
                      float* __restrict__ probs,         // [n, 256]
                      float* __restrict__ stdev,         // [n]
                      int n)
{
    __shared__ _Float16 s16[2][DIM];     // double-buffered carried state, f16
    __shared__ float    red[WAVES][4];   // per-wave partial sums (b128 rows)

    const int tid   = threadIdx.x;
    const int w     = tid >> 5;          // wave id 0..7
    const int lane  = tid & 31;
    const int lhalf = lane >> 4;         // 0: lanes 0-15, 1: lanes 16-31
    const int l16   = lane & 15;
    const int idx   = (w << 5) | lane;   // this thread's element 0..255

    // ---- One-time: gather T into B-operand layout registers (f16). ----
    // Wave w owns output blocks Q0=2w, Q1=2w+1. For block Q, K-chunk s:
    //   B[k][q] = T[16Q+q][32s+k], q = lane&15,
    //   lanes 0-15 hold klocal 0..15, lanes 16-31 hold klocal 16..31,
    //   v16h element e  <->  klocal = 16*lhalf + e  (contiguous halfs).
    v16h B[2][8];
    #pragma unroll
    for (int q = 0; q < 2; ++q) {
        const int row = (((w << 1) + q) << 4) + l16;   // = global output index
        #pragma unroll
        for (int s = 0; s < 8; ++s) {
            v16h b;
            #pragma unroll
            for (int e = 0; e < 16; ++e) {
                b[e] = (_Float16)T[row * DIM + (s << 5) + (lhalf << 4) + e];
            }
            B[q][s] = b;
        }
    }

    // bins: hr(i) = 0.5*60 + 3.0*60*i/256
    const float bin  = 30.0f + (180.0f / 256.0f) * (float)idx;
    const float bin2 = bin * bin;

    // Seed: v_{-1} = VSCALE * state0  (sum = VSCALE since state0 sums to 1)
    float vprev = state0[idx] * VSCALE;
    s16[0][idx] = (_Float16)vprev;

    // ps register prefetch pipeline
    float pq[PF];
    #pragma unroll
    for (int j = 0; j < PF; ++j) {
        int r = (j < n) ? j : (n > 0 ? n - 1 : 0);
        pq[j] = ps[(size_t)r * DIM + idx];
    }

    __syncthreads();

    int p = 0;
    for (int i = 0; i < n; ++i) {
        // ---- consume / refill ps pipeline (off-path) ----
        const float psv = pq[0];
        #pragma unroll
        for (int j = 0; j < PF - 1; ++j) pq[j] = pq[j + 1];
        int nx = i + PF;      if (nx > n - 1) nx = n - 1;
        pq[PF - 1] = ps[(size_t)nx * DIM + idx];
        int pfr = i + PF_L2;  if (pfr > n - 1) pfr = n - 1;
        __builtin_prefetch(ps + (size_t)pfr * DIM + idx, 0, 0);

        // ---- XDL pipe: matvec on v_{i-1} (from LDS buffer p) ----
        // A 16x32 f16 layout: k = 16*lhalf+e (contiguous halfs, matches B)
        v8f acc0 = {};
        v8f acc1 = {};
        #pragma unroll
        for (int s = 0; s < 8; ++s) {
            const _Float16* sp = &s16[p][s << 5];
            v8h lo = *(const v8h*)(sp + (lhalf << 4));        // ds_load_b128
            v8h hi = *(const v8h*)(sp + 8 + (lhalf << 4));    // ds_load_b128
            v16h A = __builtin_shufflevector(lo, hi,
                     0, 1, 2, 3, 4, 5, 6, 7, 8, 9, 10, 11, 12, 13, 14, 15);
            acc0 = __builtin_amdgcn_wmma_f32_16x16x32_f16(
                       false, A, false, B[0][s], (short)0, acc0, false, false);
            acc1 = __builtin_amdgcn_wmma_f32_16x16x32_f16(
                       false, A, false, B[1][s], (short)0, acc1, false, false);
        }

        // ---- DS pipe (overlaps WMMA): butterfly sums of v_{i-1} ----
        float r0 = vprev;
        float r1 = vprev * bin;
        float r2 = vprev * bin2;
        #pragma unroll
        for (int off = 16; off > 0; off >>= 1) {
            r0 += __shfl_xor(r0, off, 32);
            r1 += __shfl_xor(r1, off, 32);
            r2 += __shfl_xor(r2, off, 32);
        }
        if (lane == 0) {
            v4f t = {r0, r1, r2, 0.0f};
            *(v4f*)red[w] = t;                                // ds_store_b128
        }
        __syncthreads();   // B1: red exchange (hidden under XDL drain)

        float S0 = 0.f, S1 = 0.f, S2 = 0.f;
        #pragma unroll
        for (int k = 0; k < WAVES; ++k) {
            v4f q = *(const v4f*)red[k];                      // broadcast b128
            S0 += q.x; S1 += q.y; S2 += q.z;
        }
        const float inv = fast_rcp(S0);

        // ---- lagged outputs for sample i-1 (i==0 writes deterministic
        //      dummies to index 0, overwritten at i==1 / epilogue) ----
        const int om = (i > 0) ? (i - 1) : 0;
        probs[(size_t)om * DIM + idx] = vprev * inv;
        if (tid == 0) {
            const float Ex  = S1 * inv;
            const float Ex2 = S2 * inv;
            stdev[om] = sqrtf(Ex2 - Ex * Ex);
        }

        // ---- finish recursion: v_i = prior * ps * (VSCALE/s_{i-1}) ----
        const float prior = lhalf ? acc1[0] : acc0[0];
        const float v = prior * (psv * (VSCALE * inv));
        s16[p ^ 1][idx] = (_Float16)v;
        vprev = v;
        __syncthreads();   // B2: v_i visible; red reads done before next write
        p ^= 1;
    }

    // ---- epilogue: normalize & emit the last sample ----
    if (n > 0) {
        float r0 = vprev;
        float r1 = vprev * bin;
        float r2 = vprev * bin2;
        #pragma unroll
        for (int off = 16; off > 0; off >>= 1) {
            r0 += __shfl_xor(r0, off, 32);
            r1 += __shfl_xor(r1, off, 32);
            r2 += __shfl_xor(r2, off, 32);
        }
        if (lane == 0) {
            v4f t = {r0, r1, r2, 0.0f};
            *(v4f*)red[w] = t;
        }
        __syncthreads();
        float S0 = 0.f, S1 = 0.f, S2 = 0.f;
        #pragma unroll
        for (int k = 0; k < WAVES; ++k) {
            v4f q = *(const v4f*)red[k];
            S0 += q.x; S1 += q.y; S2 += q.z;
        }
        const float inv = fast_rcp(S0);
        probs[(size_t)(n - 1) * DIM + idx] = vprev * inv;
        if (tid == 0) {
            const float Ex  = S1 * inv;
            const float Ex2 = S2 * inv;
            stdev[n - 1] = sqrtf(Ex2 - Ex * Ex);
        }
    }
}

extern "C" void kernel_launch(void* const* d_in, const int* in_sizes, int n_in,
                              void* d_out, int out_size, void* d_ws, size_t ws_size,
                              hipStream_t stream) {
    (void)n_in; (void)out_size; (void)d_ws; (void)ws_size;
    const float* ps     = (const float*)d_in[0];   // [n,256] f32
    const float* T      = (const float*)d_in[1];   // [256,256] f32
    const float* state0 = (const float*)d_in[2];   // [256] f32
    const int n = in_sizes[0] / DIM;

    float* probs = (float*)d_out;                  // [n,256]
    float* stdev = probs + (size_t)n * DIM;        // [n]

    // Serial recurrence: one workgroup (8 wave32) owns the whole chain.
    hmm_forward_wmma<<<dim3(1), dim3(256), 0, stream>>>(ps, T, state0,
                                                        probs, stdev, n);
}
